// MultiHeadAttention_18571438588050
// MI455X (gfx1250) — matrix operand order repaired and checked by reading, not compiled, not run on hardware
//
#include <hip/hip_runtime.h>

// ---------------------------------------------------------------------------
// MHA forward for MI455X (gfx1250, wave32, WMMA).
//   S=4096, D=768, H=12, Dh=64.
// Pipeline: 3x projection GEMM (fp32->f16 convert in LDS staging, WMMA f16,
// fp32 accum) -> flash attention (WMMA f16 QK^T / PV, base-2 online softmax,
// K staged transposed in LDS, V staged via async-to-LDS) -> output projection.
// Workspace: qp/kp/vp f16 [H][S][64] + ctx f16 [S][768]  (~25.2 MB).
// ---------------------------------------------------------------------------

#define S_LEN    4096
#define DMODEL   768
#define NHEADS   12
#define HDIM     64

typedef _Float16 half16 __attribute__((ext_vector_type(16)));
typedef float    floatx8 __attribute__((ext_vector_type(8)));
typedef int      intx4   __attribute__((ext_vector_type(4)));

// gfx1250 async load-to-LDS (ASYNCcnt) if the toolchain exposes it.
// Probed prototype: (v4i addrspace(1)* src, v4i addrspace(3)* dst,
//                    imm int offset (added to BOTH addresses), imm int cpol).
#if defined(__has_builtin)
#if __has_builtin(__builtin_amdgcn_global_load_async_to_lds_b128) && \
    __has_builtin(__builtin_amdgcn_s_wait_asynccnt)
#define USE_ASYNC_LDS 1
#endif
#endif
#ifndef USE_ASYNC_LDS
#define USE_ASYNC_LDS 0
#endif

#if USE_ASYNC_LDS
#define ASYNC_B128(gp, lp, OFF)                                        \
    __builtin_amdgcn_global_load_async_to_lds_b128(                    \
        (__attribute__((address_space(1))) intx4*)(void*)(gp),         \
        (__attribute__((address_space(3))) intx4*)(void*)(lp),         \
        (OFF), 0)
#define ASYNC_WAIT() __builtin_amdgcn_s_wait_asynccnt(0)
#endif

static __device__ __forceinline__ floatx8 wmma_f16(half16 a, half16 b, floatx8 c) {
    // D = A(16x32 f16) * B(32x16 f16) + C(16x16 f32)
    return __builtin_amdgcn_wmma_f32_16x16x32_f16(false, a, false, b, (short)0, c,
                                                  false, false);
}

static __device__ __forceinline__ float rowmax16(float v) {
    v = fmaxf(v, __shfl_xor(v, 1, 32));
    v = fmaxf(v, __shfl_xor(v, 2, 32));
    v = fmaxf(v, __shfl_xor(v, 4, 32));
    v = fmaxf(v, __shfl_xor(v, 8, 32));
    return v;
}
static __device__ __forceinline__ float rowsum16(float v) {
    v += __shfl_xor(v, 1, 32);
    v += __shfl_xor(v, 2, 32);
    v += __shfl_xor(v, 4, 32);
    v += __shfl_xor(v, 8, 32);
    return v;
}

// ---------------------------------------------------------------------------
// Projection GEMM:  Out[h][s][d] = f16( X[s][:] . W[h*64+d][:] + bias )
// Block tile: 128(M) x 64(N), K stepped by 32. 8 waves: 4(M) x 2(N), each
// wave 32x32 via 2x2 WMMA tiles. grid = (4096/128, 768/64=NHEADS).
// ---------------------------------------------------------------------------
__global__ __launch_bounds__(256)
void proj_kernel(const float* __restrict__ X, const float* __restrict__ W,
                 const float* __restrict__ bias, _Float16* __restrict__ OutHeads)
{
    __shared__ _Float16 lA[128 * 40];  // A tile, row stride 40 (pad)
    __shared__ _Float16 lB[32 * 72];   // B tile [k][n], row stride 72 (pad)

    const int tid   = threadIdx.x;
    const int lane  = tid & 31;
    const int wave  = tid >> 5;
    const int waveM = wave & 3;
    const int waveN = wave >> 2;
    const int m0    = blockIdx.x * 128;
    const int n0    = blockIdx.y * 64;   // == head * 64

    floatx8 acc[2][2];
    {
        floatx8 z = {};
        acc[0][0] = z; acc[0][1] = z; acc[1][0] = z; acc[1][1] = z;
    }

    for (int k0 = 0; k0 < DMODEL; k0 += 32) {
        __syncthreads();
        // Stage A: 128x32 fp32 -> f16
        {
            const int row = tid >> 1;
            const int col = (tid & 1) * 16;
            const float* src = X + (size_t)(m0 + row) * DMODEL + k0 + col;
            _Float16* dst = &lA[row * 40 + col];
            #pragma unroll
            for (int i = 0; i < 16; ++i) dst[i] = (_Float16)src[i];
            if (k0 + 32 < DMODEL) __builtin_prefetch(src + 32, 0, 3);
        }
        // Stage B (transposed): lB[k][n] = W[n0+n][k0+k]
        {
            const int n  = tid >> 2;
            const int ks = (tid & 3) * 8;
            const float* src = W + (size_t)(n0 + n) * DMODEL + k0 + ks;
            #pragma unroll
            for (int j = 0; j < 8; ++j) lB[(ks + j) * 72 + n] = (_Float16)src[j];
            if (k0 + 32 < DMODEL) __builtin_prefetch(src + 32, 0, 3);
        }
        __syncthreads();

        // A fragments: lane=M row, elems 0..7 -> K kb..kb+7, 8..15 -> K 16+kb..
        half16 af[2];
        {
            const int kb = (lane >> 4) * 8;
            #pragma unroll
            for (int mi = 0; mi < 2; ++mi) {
                const _Float16* p = &lA[(waveM * 32 + mi * 16 + (lane & 15)) * 40];
                #pragma unroll
                for (int i = 0; i < 8; ++i) {
                    af[mi][i]     = p[kb + i];
                    af[mi][8 + i] = p[16 + kb + i];
                }
            }
        }
        // B fragments: lane=N, elems = K (same order as A)
        half16 bf[2];
        #pragma unroll
        for (int ni = 0; ni < 2; ++ni) {
            const _Float16* p = &lB[waveN * 32 + ni * 16 + (lane & 15)];
            #pragma unroll
            for (int n = 0; n < 16; ++n) bf[ni][n] = p[((n & 7) + (lane >> 4) * 8 + (n >> 3) * 16) * 72];
        }
        #pragma unroll
        for (int mi = 0; mi < 2; ++mi)
            #pragma unroll
            for (int ni = 0; ni < 2; ++ni)
                acc[mi][ni] = wmma_f16(af[mi], bf[ni], acc[mi][ni]);
    }

    // Epilogue: add bias, write f16 to [h][s][d]
    const int h       = blockIdx.y;
    const int rowHalf = (lane >> 4) * 8;   // M offset 0 or 8
    const int ncol    = lane & 15;
    #pragma unroll
    for (int mi = 0; mi < 2; ++mi) {
        #pragma unroll
        for (int ni = 0; ni < 2; ++ni) {
            const int d = waveN * 32 + ni * 16 + ncol;
            const float b = bias[n0 + d];
            #pragma unroll
            for (int r = 0; r < 8; ++r) {
                const int m = m0 + waveM * 32 + mi * 16 + rowHalf + r;
                OutHeads[(size_t)h * (S_LEN * HDIM) + (size_t)m * HDIM + d] =
                    (_Float16)(acc[mi][ni][r] + b);
            }
        }
    }
}

// ---------------------------------------------------------------------------
// Flash attention. grid = (S/128, NHEADS), block = 256 (8 waves).
// Each wave: 16 query rows, full 64-dim fp32 accumulator, online softmax.
// K tile staged TRANSPOSED (lkT[d][key]); score B-fragment elements are read
// at a 72-half LDS stride (one d row each); V tile staged with async load-to-LDS when available.
// ---------------------------------------------------------------------------
__global__ __launch_bounds__(256)
void attn_kernel(const _Float16* __restrict__ Qh, const _Float16* __restrict__ Kh,
                 const _Float16* __restrict__ Vh, _Float16* __restrict__ Ctx)
{
    __shared__ _Float16 lkT[64 * 72];         // K tile transposed: [d][key]
    __shared__ _Float16 lv[64 * 72];          // V tile [key][d]
    __shared__ _Float16 pbuf[8 * 16 * 72];    // per-wave P re-layout scratch

    const int tid  = threadIdx.x;
    const int lane = tid & 31;
    const int wave = tid >> 5;
    const int h    = blockIdx.y;
    const int q0   = blockIdx.x * 128 + wave * 16;

    const _Float16* Qbase = Qh + (size_t)h * S_LEN * HDIM;
    const _Float16* Kbase = Kh + (size_t)h * S_LEN * HDIM;
    const _Float16* Vbase = Vh + (size_t)h * S_LEN * HDIM;

    // Q A-fragments for the two 32-wide K-chunks of the 64-dim head
    half16 aQ[2];
    {
        const int kb = (lane >> 4) * 8;
        const _Float16* p = Qbase + (size_t)(q0 + (lane & 15)) * HDIM;
        #pragma unroll
        for (int ks = 0; ks < 2; ++ks)
            #pragma unroll
            for (int i = 0; i < 8; ++i) {
                aQ[ks][i]     = p[ks * 32 + kb + i];
                aQ[ks][8 + i] = p[ks * 32 + 16 + kb + i];
            }
    }

    floatx8 accO[4];
    { floatx8 z = {}; accO[0] = z; accO[1] = z; accO[2] = z; accO[3] = z; }
    float m_run[8], l_run[8];
    #pragma unroll
    for (int r = 0; r < 8; ++r) { m_run[r] = -1.0e30f; l_run[r] = 0.0f; }

    // scores scaled into base-2 domain: 1/sqrt(64) * log2(e)
    const float SCALE2 = 0.125f * 1.44269504088896340736f;

    _Float16* pw = &pbuf[wave * 16 * 72];

    for (int key0 = 0; key0 < S_LEN; key0 += 64) {
        __syncthreads();
        {
            const int row = tid >> 2;        // key 0..63
            const int col = (tid & 3) * 16;  // d chunk
            // V tile: straight row-major copy (16 halves / thread)
            const _Float16* vsrc = Vbase + (size_t)(key0 + row) * HDIM + col;
#if USE_ASYNC_LDS
            ASYNC_B128(vsrc, &lv[row * 72 + col], 0);
            ASYNC_B128(vsrc, &lv[row * 72 + col], 16);
#else
            {
                const int4* s = (const int4*)vsrc;
                int4* d = (int4*)&lv[row * 72 + col];
                d[0] = s[0]; d[1] = s[1];
            }
#endif
            // K tile: load row, scatter transposed into lkT[d][key]
            const _Float16* ksrc = (const _Float16*)__builtin_assume_aligned(
                Kbase + (size_t)(key0 + row) * HDIM + col, 16);
            #pragma unroll
            for (int i = 0; i < 16; ++i) lkT[(col + i) * 72 + row] = ksrc[i];

            if (key0 + 64 < S_LEN) {
                __builtin_prefetch(ksrc + 64 * HDIM, 0, 3);
                __builtin_prefetch(vsrc + 64 * HDIM, 0, 3);
            }
        }
#if USE_ASYNC_LDS
        ASYNC_WAIT();
#endif
        __syncthreads();

        // Scores: S(16x64) = Q(16x64) . K^T  -> 4 N-tiles x 2 K-chunks WMMA
        float sc[4][8];
        #pragma unroll
        for (int nt = 0; nt < 4; ++nt) {
            floatx8 c = {};
            #pragma unroll
            for (int ks = 0; ks < 2; ++ks) {
                half16 bk;  // lane = key column, elems = d-within-chunk
                const _Float16* p = &lkT[(ks * 32) * 72 + nt * 16 + (lane & 15)];
                #pragma unroll
                for (int n = 0; n < 16; ++n) bk[n] = p[((n & 7) + (lane >> 4) * 8 + (n >> 3) * 16) * 72];
                c = wmma_f16(aQ[ks], bk, c);
            }
            #pragma unroll
            for (int r = 0; r < 8; ++r) sc[nt][r] = c[r] * SCALE2;
        }

        // Online softmax per row (base-2 domain); row r lives on 16 lanes
        #pragma unroll
        for (int r = 0; r < 8; ++r) {
            float mx = fmaxf(fmaxf(sc[0][r], sc[1][r]), fmaxf(sc[2][r], sc[3][r]));
            mx = rowmax16(mx);
            const float mnew  = fmaxf(m_run[r], mx);
            const float alpha = exp2f(m_run[r] - mnew);
            float psum = 0.0f;
            #pragma unroll
            for (int nt = 0; nt < 4; ++nt) {
                const float p = exp2f(sc[nt][r] - mnew);
                sc[nt][r] = p;
                psum += p;
            }
            psum = rowsum16(psum);
            l_run[r] = l_run[r] * alpha + psum;
            m_run[r] = mnew;
            #pragma unroll
            for (int nt = 0; nt < 4; ++nt) accO[nt][r] *= alpha;
        }

        // Re-layout P: C-fragment (lane=key col) -> LDS -> A-fragment (lane=row)
        {
            const int Mrow = (lane >> 4) * 8;
            const int ncol = lane & 15;
            #pragma unroll
            for (int r = 0; r < 8; ++r)
                #pragma unroll
                for (int nt = 0; nt < 4; ++nt)
                    pw[(Mrow + r) * 72 + nt * 16 + ncol] = (_Float16)sc[nt][r];
        }
        __builtin_amdgcn_wave_barrier();  // keep LDS write -> read ordered

        half16 aP[2];
        {
            const int kb = (lane >> 4) * 8;
            const _Float16* p = &pw[(lane & 15) * 72];
            #pragma unroll
            for (int ks = 0; ks < 2; ++ks)
                #pragma unroll
                for (int i = 0; i < 8; ++i) {
                    aP[ks][i]     = p[ks * 32 + kb + i];
                    aP[ks][8 + i] = p[ks * 32 + 16 + kb + i];
                }
        }

        // O += P(16x64) . V(64x64)
        #pragma unroll
        for (int nt = 0; nt < 4; ++nt)
            #pragma unroll
            for (int ks = 0; ks < 2; ++ks) {
                half16 bv;  // lane = d column, elems = key row within chunk
                const _Float16* p = &lv[(ks * 32) * 72 + nt * 16 + (lane & 15)];
                #pragma unroll
                for (int n = 0; n < 16; ++n) bv[n] = p[((n & 7) + (lane >> 4) * 8 + (n >> 3) * 16) * 72];
                accO[nt] = wmma_f16(aP[ks], bv, accO[nt]);
            }
    }

    // Normalize and write ctx[s][h*64+d] as f16
    {
        const int Mrow = (lane >> 4) * 8;
        const int ncol = lane & 15;
        #pragma unroll
        for (int r = 0; r < 8; ++r) {
            const int   srow = q0 + Mrow + r;
            const float inv  = 1.0f / l_run[r];
            #pragma unroll
            for (int nt = 0; nt < 4; ++nt) {
                const int d = nt * 16 + ncol;
                Ctx[(size_t)srow * DMODEL + h * HDIM + d] =
                    (_Float16)(accO[nt][r] * inv);
            }
        }
    }
}

// ---------------------------------------------------------------------------
// Output projection: Out[s][o] = ctx[s][:] . Wo[o][:] + bo  (fp32 out).
// Same tiling as proj_kernel; A already f16 (async-staged when available).
// ---------------------------------------------------------------------------
__global__ __launch_bounds__(256)
void out_proj_kernel(const _Float16* __restrict__ Ctx, const float* __restrict__ W,
                     const float* __restrict__ bias, float* __restrict__ Out)
{
    __shared__ _Float16 lA[128 * 40];
    __shared__ _Float16 lB[32 * 72];

    const int tid   = threadIdx.x;
    const int lane  = tid & 31;
    const int wave  = tid >> 5;
    const int waveM = wave & 3;
    const int waveN = wave >> 2;
    const int m0    = blockIdx.x * 128;
    const int n0    = blockIdx.y * 64;

    floatx8 acc[2][2];
    {
        floatx8 z = {};
        acc[0][0] = z; acc[0][1] = z; acc[1][0] = z; acc[1][1] = z;
    }

    for (int k0 = 0; k0 < DMODEL; k0 += 32) {
        __syncthreads();
        // Stage A: straight f16 copy (16 halves / thread)
        {
            const int row = tid >> 1;
            const int col = (tid & 1) * 16;
            const _Float16* src = Ctx + (size_t)(m0 + row) * DMODEL + k0 + col;
#if USE_ASYNC_LDS
            ASYNC_B128(src, &lA[row * 40 + col], 0);
            ASYNC_B128(src, &lA[row * 40 + col], 16);
#else
            {
                const int4* s = (const int4*)src;
                int4* d = (int4*)&lA[row * 40 + col];
                d[0] = s[0]; d[1] = s[1];
            }
#endif
            if (k0 + 32 < DMODEL) __builtin_prefetch(src + 32, 0, 3);
        }
        // Stage B transposed from fp32 Wo
        {
            const int n  = tid >> 2;
            const int ks = (tid & 3) * 8;
            const float* src = W + (size_t)(n0 + n) * DMODEL + k0 + ks;
            #pragma unroll
            for (int j = 0; j < 8; ++j) lB[(ks + j) * 72 + n] = (_Float16)src[j];
            if (k0 + 32 < DMODEL) __builtin_prefetch(src + 32, 0, 3);
        }
#if USE_ASYNC_LDS
        ASYNC_WAIT();
#endif
        __syncthreads();

        half16 af[2];
        {
            const int kb = (lane >> 4) * 8;
            #pragma unroll
            for (int mi = 0; mi < 2; ++mi) {
                const _Float16* p = &lA[(waveM * 32 + mi * 16 + (lane & 15)) * 40];
                #pragma unroll
                for (int i = 0; i < 8; ++i) {
                    af[mi][i]     = p[kb + i];
                    af[mi][8 + i] = p[16 + kb + i];
                }
            }
        }
        half16 bf[2];
        #pragma unroll
        for (int ni = 0; ni < 2; ++ni) {
            const _Float16* p = &lB[waveN * 32 + ni * 16 + (lane & 15)];
            #pragma unroll
            for (int n = 0; n < 16; ++n) bf[ni][n] = p[((n & 7) + (lane >> 4) * 8 + (n >> 3) * 16) * 72];
        }
        #pragma unroll
        for (int mi = 0; mi < 2; ++mi)
            #pragma unroll
            for (int ni = 0; ni < 2; ++ni)
                acc[mi][ni] = wmma_f16(af[mi], bf[ni], acc[mi][ni]);
    }

    const int rowHalf = (lane >> 4) * 8;
    const int ncol    = lane & 15;
    #pragma unroll
    for (int mi = 0; mi < 2; ++mi) {
        #pragma unroll
        for (int ni = 0; ni < 2; ++ni) {
            const int n = n0 + waveN * 32 + ni * 16 + ncol;
            const float b = bias[n];
            #pragma unroll
            for (int r = 0; r < 8; ++r) {
                const int m = m0 + waveM * 32 + mi * 16 + rowHalf + r;
                Out[(size_t)m * DMODEL + n] = acc[mi][ni][r] + b;
            }
        }
    }
}

// ---------------------------------------------------------------------------
extern "C" void kernel_launch(void* const* d_in, const int* in_sizes, int n_in,
                              void* d_out, int out_size, void* d_ws, size_t ws_size,
                              hipStream_t stream) {
    (void)in_sizes; (void)n_in; (void)out_size; (void)ws_size;

    const float* q  = (const float*)d_in[0];
    const float* k  = (const float*)d_in[1];
    const float* v  = (const float*)d_in[2];
    const float* Wq = (const float*)d_in[3];
    const float* bq = (const float*)d_in[4];
    const float* Wk = (const float*)d_in[5];
    const float* bk = (const float*)d_in[6];
    const float* Wv = (const float*)d_in[7];
    const float* bv = (const float*)d_in[8];
    const float* Wo = (const float*)d_in[9];
    const float* bo = (const float*)d_in[10];

    const size_t headElems = (size_t)NHEADS * S_LEN * HDIM;  // 3,145,728
    _Float16* qp  = (_Float16*)d_ws;
    _Float16* kp  = qp + headElems;
    _Float16* vp  = kp + headElems;
    _Float16* ctx = vp + headElems;   // [S][DMODEL] f16; total ws ~25.2 MB

    dim3 block(256);
    dim3 gridP(S_LEN / 128, NHEADS);      // (32, 12)
    dim3 gridA(S_LEN / 128, NHEADS);      // (32, 12)
    dim3 gridO(S_LEN / 128, DMODEL / 64); // (32, 12)

    proj_kernel<<<gridP, block, 0, stream>>>(q, Wq, bq, qp);
    proj_kernel<<<gridP, block, 0, stream>>>(k, Wk, bk, kp);
    proj_kernel<<<gridP, block, 0, stream>>>(v, Wv, bv, vp);
    attn_kernel<<<gridA, block, 0, stream>>>(qp, kp, vp, ctx);
    out_proj_kernel<<<gridO, block, 0, stream>>>(ctx, Wo, bo, (float*)d_out);
}
